// Mamba2MIL_27453430956108
// MI455X (gfx1250) — matrix operand order repaired and checked by reading, not compiled, not run on hardware
//
#include <hip/hip_runtime.h>
#include <hip/hip_bf16.h>

typedef __attribute__((ext_vector_type(16))) _Float16     v16h;
typedef __attribute__((ext_vector_type(8)))  float        v8f;
typedef __attribute__((ext_vector_type(4)))  unsigned int v4u;
typedef __attribute__((ext_vector_type(8)))  int          v8i;
typedef __attribute__((ext_vector_type(4)))  int          v4i;

#define DMODEL  512
#define DINNER  2048
#define DXBC    2560
#define DSTATE  256
#define NH      32
#define PDIM    64
#define SEQL    4096
#define NB      2
#define NTOK    (NB * SEQL)

// ---------------------------------------------------------------------------
// Tensor Data Mover: issue a 2D tile load (f32 elements) into LDS.
// D# group0: count=1 | lds_addr | global_addr(57b) | type=2
// D# group1: wg_mask=0, data_size=2(4B), tensor_dim0/1, tile_dim0/1, stride0
// Wave-level op (EXEC ignored) -> call under a wave-uniform guard only.
// ---------------------------------------------------------------------------
__device__ __forceinline__ unsigned lds_off(const void* p) {
    return (unsigned)(unsigned long long)(uintptr_t)p;   // low 32b = LDS offset
}

__device__ __forceinline__ void tdm_load_2d_f32(const float* gsrc, unsigned ldsoff,
                                                unsigned tile_x, unsigned tile_y,
                                                unsigned dim0, unsigned dim1,
                                                unsigned long long stride0) {
    unsigned long long ga = (unsigned long long)(uintptr_t)gsrc;
    v4u g0;
    g0.x = 1u;                                            // count=1, user mode
    g0.y = ldsoff;                                        // lds_addr
    g0.z = (unsigned)(ga & 0xffffffffull);                // global_addr lo
    g0.w = (unsigned)((ga >> 32) & 0x01ffffffull) | (2u << 30);  // addr hi | type=2
    v8i g1;
    g1[0] = (int)(2u << 16);                              // data_size = 4 bytes
    g1[1] = (int)((dim0 & 0xffffu) << 16);                // tensor_dim0 lo16
    g1[2] = (int)(((dim0 >> 16) & 0xffffu) | ((dim1 & 0xffffu) << 16));
    g1[3] = (int)(((dim1 >> 16) & 0xffffu) | ((tile_x & 0xffffu) << 16)); // tile_dim0
    g1[4] = (int)(tile_y & 0xffffu);                      // tile_dim1 (tile_dim2=0)
    g1[5] = (int)(unsigned)(stride0 & 0xffffffffull);     // tensor_dim0_stride lo
    g1[6] = (int)(unsigned)((stride0 >> 32) & 0xffffull); // stride hi16 | stride1=0
    g1[7] = 0;
    v4i z4 = {0, 0, 0, 0};
#if __clang_major__ >= 23
    v8i z8 = {0, 0, 0, 0, 0, 0, 0, 0};
    __builtin_amdgcn_tensor_load_to_lds(g0, g1, z4, z4, z8, 0);
#else
    __builtin_amdgcn_tensor_load_to_lds(g0, g1, z4, z4, 0);
#endif
}

// ---------------------------------------------------------------------------
// WMMA GEMM:  C[M,N] = act( A[M,K] @ W[N,K]^T + bias )
// Block: 8 waves; block tile 16(M) x 512(N); each wave owns a 16x64 strip
// (4 accumulators sharing one A fragment -> 4 wmma per A-fragment read).
// A K-slice (16x32 f32) is TDM-staged to LDS (wave 0 issues, double buffered,
// s_wait_tensorcnt + barrier); all 8 waves read A fragments from LDS.
// Tile guards are readfirstlane'd -> scalar branches, EXEC stays all-ones
// around every WMMA (ISA requirement). ACT: 0=none, 1=relu, 2=tanh
// ---------------------------------------------------------------------------
template<int ACT>
__global__ __launch_bounds__(256)
void wmma_gemm_kernel(const float* __restrict__ Aact,
                      const float* __restrict__ W,
                      const float* __restrict__ bias,
                      float* __restrict__ Cout,
                      int M, int N, int K)
{
    __shared__ float Atile[2][16 * 32];
    const int lane = threadIdx.x;            // 0..31
    const int widu = __builtin_amdgcn_readfirstlane((int)threadIdx.y); // scalar
    const int m0   = blockIdx.y * 16;
    const int n0   = __builtin_amdgcn_readfirstlane(
                         (int)(blockIdx.x * 512 + widu * 64));          // scalar

    const int mr    = lane & 15;
    const int khalf = (lane >> 4) * 8;       // unused (0 or 8); A fragment also uses kbh
    const int kbh   = (lane >> 4) * 16;      // B K sub-group: 0 or 16

    int nc[4];
    const float* wr[4];
    bool ok[4];
#pragma unroll
    for (int j = 0; j < 4; ++j) {
        ok[j] = (n0 + 16 * j) < N;           // scalar per wave
        nc[j] = n0 + 16 * j + mr;
        wr[j] = W + (size_t)(ok[j] ? nc[j] : 0) * K;
    }

    const int nk = K / 32;
    if (widu == 0)
        tdm_load_2d_f32(Aact + (size_t)m0 * K, lds_off(&Atile[0][0]),
                        32, 16, (unsigned)K, (unsigned)M, (unsigned long long)K);

    v8f acc[4] = {};
    for (int i = 0; i < nk; ++i) {
        const int cur = i & 1;
        if (widu == 0) {
            if (i + 1 < nk) {
                tdm_load_2d_f32(Aact + (size_t)m0 * K + (size_t)(i + 1) * 32,
                                lds_off(&Atile[1 - cur][0]),
                                32, 16, (unsigned)K, (unsigned)M,
                                (unsigned long long)K);
                __builtin_amdgcn_s_wait_tensorcnt(1);   // slice i has landed
            } else {
                __builtin_amdgcn_s_wait_tensorcnt(0);
            }
        }
        __syncthreads();                     // publish Atile[cur]

        // A fragment from LDS (contiguous K: k = 16*(lane>>4) + s, matches B)
        const float* at = &Atile[cur][mr * 32];
        v16h af;
#pragma unroll
        for (int j = 0; j < 8; ++j) {
            af[j]     = (_Float16)at[kbh + j];
            af[j + 8] = (_Float16)at[kbh + 8 + j];
        }
        const int k0 = i * 32;
#pragma unroll
        for (int j = 0; j < 4; ++j) {
            if (ok[j]) {
                const float* bp = wr[j] + k0 + kbh;
                __builtin_prefetch((const void*)(bp + 32), 0, 1);
                v16h bf;
#pragma unroll
                for (int q = 0; q < 16; ++q) bf[q] = (_Float16)bp[q];
                acc[j] = __builtin_amdgcn_wmma_f32_16x16x32_f16(
                    false, af, false, bf, (short)0, acc[j], false, false);
            }
        }
        __syncthreads();                     // all reads of Atile[cur] done
    }

    // C/D layout: lane -> N ; VGPR r -> M = r + (lane>=16 ? 8 : 0)
    const int rbase = (lane >> 4) * 8;
#pragma unroll
    for (int j = 0; j < 4; ++j) {
        if (ok[j]) {
            const float bv = bias ? bias[nc[j]] : 0.0f;
#pragma unroll
            for (int r = 0; r < 8; ++r) {
                float v = acc[j][r] + bv;
                if (ACT == 1) v = fmaxf(v, 0.0f);
                if (ACT == 2) v = tanhf(v);
                Cout[(size_t)(m0 + rbase + r) * N + nc[j]] = v;
            }
        }
    }
}

// ---------------------------------------------------------------------------
// Row LayerNorm (D=512), one block (256 thr) per row
// ---------------------------------------------------------------------------
__global__ void layernorm_kernel(const float* __restrict__ in,
                                 float* __restrict__ out,
                                 const float* __restrict__ w,
                                 const float* __restrict__ b, int D)
{
    const int row = blockIdx.x, tid = threadIdx.x;
    const float* x = in + (size_t)row * D;
    __shared__ float red[8];
    float s = 0.f;
    for (int i = tid; i < D; i += 256) s += x[i];
    for (int off = 16; off > 0; off >>= 1) s += __shfl_xor(s, off);
    if ((tid & 31) == 0) red[tid >> 5] = s;
    __syncthreads();
    if (tid < 8) { float v = red[tid];
        for (int off = 4; off > 0; off >>= 1) v += __shfl_xor(v, off);
        if (tid == 0) red[0] = v; }
    __syncthreads();
    const float mean = red[0] / (float)D;
    __syncthreads();
    float vs = 0.f;
    for (int i = tid; i < D; i += 256) { float d = x[i] - mean; vs += d * d; }
    for (int off = 16; off > 0; off >>= 1) vs += __shfl_xor(vs, off);
    if ((tid & 31) == 0) red[tid >> 5] = vs;
    __syncthreads();
    if (tid < 8) { float v = red[tid];
        for (int off = 4; off > 0; off >>= 1) v += __shfl_xor(v, off);
        if (tid == 0) red[0] = v; }
    __syncthreads();
    const float rstd = rsqrtf(red[0] / (float)D + 1e-5f);
    float* op = out + (size_t)row * D;
    for (int i = tid; i < D; i += 256) op[i] = (x[i] - mean) * rstd * w[i] + b[i];
}

// ---------------------------------------------------------------------------
// Depthwise causal conv (width 4) + bias + SiLU over xBC channels
// ---------------------------------------------------------------------------
__global__ void dwconv_silu_kernel(const float* __restrict__ xin,
                                   const float* __restrict__ w,
                                   const float* __restrict__ bias,
                                   float* __restrict__ out, int L)
{
    const size_t idx = (size_t)blockIdx.x * 256 + threadIdx.x;
    const int c = (int)(idx % DXBC);
    const int t = (int)((idx / DXBC) % L);
    const int b = (int)(idx / ((size_t)DXBC * L));
    float acc = bias[c];
#pragma unroll
    for (int k = 0; k < 4; ++k) {
        int ts = t - 3 + k;
        if (ts >= 0) acc = fmaf(xin[((size_t)(b * L + ts)) * DXBC + c], w[c * 4 + k], acc);
    }
    out[idx] = acc / (1.0f + __expf(-acc));        // silu
}

// ---------------------------------------------------------------------------
// Selective scan: one block per (batch, head). State 64x256 in registers
// (64 f32/thread). B/C (512 f32) and x (64 f32) rows are TDM-DMA'd into
// double-buffered LDS: wave 0 issues loads for t+1 while t is computed.
// thread tid: p = tid>>2, n-chunk = (tid&3)*64
// ---------------------------------------------------------------------------
__global__ __launch_bounds__(256)
void scan_kernel(const float* __restrict__ dtraw,   // [NTOK, NH]
                 const float* __restrict__ xbcc,    // [NTOK, DXBC] (conv'd)
                 const float* __restrict__ A_log,
                 const float* __restrict__ Dparam,
                 const float* __restrict__ dt_bias,
                 float* __restrict__ y,             // [NTOK, DINNER]
                 int L)
{
    const int b   = blockIdx.x >> 5;
    const int h   = blockIdx.x & 31;
    const int tid = threadIdx.x;              // 0..255
    const int widu = __builtin_amdgcn_readfirstlane(tid >> 5);  // scalar wave id
    const int p   = tid >> 2;
    const int nc  = (tid & 3) << 6;
    const float Ah  = -__expf(A_log[h]);
    const float Dh  = Dparam[h];
    const float dtb = dt_bias[h];
    __shared__ float BC[2][2 * DSTATE];       // B row then C row
    __shared__ float xs[2][PDIM];
    float st[64];
#pragma unroll
    for (int i = 0; i < 64; ++i) st[i] = 0.f;

    if (widu == 0) {
        const float* row0 = xbcc + (size_t)(b * L) * DXBC;
        tdm_load_2d_f32(row0 + DINNER, lds_off(&BC[0][0]), 512, 1, 512, 1, 512ull);
        tdm_load_2d_f32(row0 + h * PDIM, lds_off(&xs[0][0]), 64, 1, 64, 1, 64ull);
    }

    for (int t = 0; t < L; ++t) {
        const int cur = t & 1;
        if (widu == 0) {
            if (t + 1 < L) {
                const float* rown = xbcc + (size_t)(b * L + t + 1) * DXBC;
                tdm_load_2d_f32(rown + DINNER, lds_off(&BC[1 - cur][0]),
                                512, 1, 512, 1, 512ull);
                tdm_load_2d_f32(rown + h * PDIM, lds_off(&xs[1 - cur][0]),
                                64, 1, 64, 1, 64ull);
                __builtin_amdgcn_s_wait_tensorcnt(2);   // pair t has landed
            } else {
                __builtin_amdgcn_s_wait_tensorcnt(0);
            }
        }
        __syncthreads();

        const size_t tok = (size_t)(b * L + t);
        float dv = dtraw[tok * NH + h] + dtb;
        dv = (dv > 20.f) ? dv : log1pf(__expf(dv));     // softplus
        const float decay = __expf(dv * Ah);
        const float coeff = dv * xs[cur][p];
        const float* Bs = &BC[cur][0];
        const float* Cs = &BC[cur][DSTATE];
        float acc = 0.f;
#pragma unroll
        for (int i = 0; i < 64; ++i) {
            st[i] = fmaf(coeff, Bs[nc + i], decay * st[i]);
            acc   = fmaf(st[i], Cs[nc + i], acc);
        }
        acc += __shfl_xor(acc, 1);
        acc += __shfl_xor(acc, 2);
        if ((tid & 3) == 0)
            y[tok * DINNER + h * PDIM + p] = acc + Dh * xs[cur][p];
        __syncthreads();
    }
}

// ---------------------------------------------------------------------------
// Gated RMSNorm: out = (y*silu(z)) * rsqrt(mean((y*silu(z))^2)+eps) * w
// ---------------------------------------------------------------------------
__global__ void gated_rmsnorm_kernel(const float* __restrict__ y,
                                     const float* __restrict__ z,
                                     const float* __restrict__ w,
                                     float* __restrict__ out)
{
    const int row = blockIdx.x, tid = threadIdx.x;
    const size_t base = (size_t)row * DINNER;
    __shared__ float red[8];
    float v[8]; float ss = 0.f;
#pragma unroll
    for (int j = 0; j < 8; ++j) {
        int i = tid + j * 256;
        float zv = z[base + i];
        float g  = zv / (1.f + __expf(-zv));
        float t  = y[base + i] * g;
        v[j] = t; ss = fmaf(t, t, ss);
    }
    for (int off = 16; off > 0; off >>= 1) ss += __shfl_xor(ss, off);
    if ((tid & 31) == 0) red[tid >> 5] = ss;
    __syncthreads();
    if (tid < 8) { float t = red[tid];
        for (int off = 4; off > 0; off >>= 1) t += __shfl_xor(t, off);
        if (tid == 0) red[0] = t; }
    __syncthreads();
    const float scale = rsqrtf(red[0] / (float)DINNER + 1e-5f);
#pragma unroll
    for (int j = 0; j < 8; ++j) {
        int i = tid + j * 256;
        out[base + i] = v[j] * scale * w[i];
    }
}

// residual + mask: h = (o + h) * mask[token]
__global__ void residual_mask_kernel(const float* __restrict__ o,
                                     float* __restrict__ h,
                                     const float* __restrict__ mask, int D)
{
    const size_t idx = (size_t)blockIdx.x * 256 + threadIdx.x;
    const size_t tok = idx / D;
    h[idx] = (o[idx] + h[idx]) * mask[tok];
}

// a[tok] = dot(tproj[tok,0:128], w2) + b2   (tproj already tanh'd)
__global__ void att_score_kernel(const float* __restrict__ tproj,
                                 const float* __restrict__ w2,
                                 const float* __restrict__ b2,
                                 float* __restrict__ a)
{
    const int tok = blockIdx.x, tid = threadIdx.x;   // 128 threads
    __shared__ float red[4];
    float v = tproj[(size_t)tok * 128 + tid] * w2[tid];
    for (int off = 16; off > 0; off >>= 1) v += __shfl_xor(v, off);
    if ((tid & 31) == 0) red[tid >> 5] = v;
    __syncthreads();
    if (tid == 0) a[tok] = red[0] + red[1] + red[2] + red[3] + b2[0];
}

// masked softmax over L per batch; one block per batch
__global__ void masked_softmax_kernel(const float* __restrict__ a,
                                      const float* __restrict__ mask,
                                      float* __restrict__ Aout, int L)
{
    const int b = blockIdx.x, tid = threadIdx.x;     // 256 threads
    __shared__ float red[8];
    float mx = -1e30f;
    for (int i = tid; i < L; i += 256) {
        float v = (mask[b * L + i] == 0.f) ? -1e9f : a[b * L + i];
        mx = fmaxf(mx, v);
    }
    for (int off = 16; off > 0; off >>= 1) mx = fmaxf(mx, __shfl_xor(mx, off));
    if ((tid & 31) == 0) red[tid >> 5] = mx;
    __syncthreads();
    if (tid < 8) { float t = red[tid];
        for (int off = 4; off > 0; off >>= 1) t = fmaxf(t, __shfl_xor(t, off));
        if (tid == 0) red[0] = t; }
    __syncthreads();
    const float m = red[0];
    __syncthreads();
    float s = 0.f;
    for (int i = tid; i < L; i += 256) {
        float v = (mask[b * L + i] == 0.f) ? -1e9f : a[b * L + i];
        s += __expf(v - m);
    }
    for (int off = 16; off > 0; off >>= 1) s += __shfl_xor(s, off);
    if ((tid & 31) == 0) red[tid >> 5] = s;
    __syncthreads();
    if (tid < 8) { float t = red[tid];
        for (int off = 4; off > 0; off >>= 1) t += __shfl_xor(t, off);
        if (tid == 0) red[0] = t; }
    __syncthreads();
    const float inv = 1.f / red[0];
    for (int i = tid; i < L; i += 256) {
        float v = (mask[b * L + i] == 0.f) ? -1e9f : a[b * L + i];
        Aout[b * L + i] = __expf(v - m) * inv;
    }
}

// hpool[b,d] = sum_n A[b,n]*hf[b,n,d] ; grid = 4 (b x d-half), 256 thr
__global__ void pool_kernel(const float* __restrict__ Aw,
                            const float* __restrict__ hf,
                            float* __restrict__ hpool, int L)
{
    const int b = blockIdx.x >> 1;
    const int d = ((blockIdx.x & 1) << 8) + threadIdx.x;
    float acc = 0.f;
    for (int n = 0; n < L; ++n)
        acc = fmaf(Aw[b * L + n], hf[((size_t)(b * L + n)) * DMODEL + d], acc);
    hpool[b * DMODEL + d] = acc;
}

// classifier + softmax + argmax; 1 block x 32 thr; out = d_out base
__global__ void head_kernel(const float* __restrict__ hpool,
                            const float* __restrict__ cw,
                            const float* __restrict__ cb,
                            float* __restrict__ out)
{
    __shared__ float lg[8];
    const int tid = threadIdx.x;
    if (tid < 8) {
        int b = tid >> 2, c = tid & 3;
        float acc = cb[c];
        for (int i = 0; i < DMODEL; ++i)
            acc = fmaf(hpool[b * DMODEL + i], cw[c * DMODEL + i], acc);
        lg[tid] = acc;
        out[tid] = acc;                               // logits
    }
    __syncthreads();
    if (tid < 2) {
        int b = tid;
        float m = lg[b * 4]; int am = 0;
        for (int c = 1; c < 4; ++c) if (lg[b * 4 + c] > m) { m = lg[b * 4 + c]; am = c; }
        float s = 0.f;
        for (int c = 0; c < 4; ++c) s += __expf(lg[b * 4 + c] - m);
        for (int c = 0; c < 4; ++c) out[8 + b * 4 + c] = __expf(lg[b * 4 + c] - m) / s;
        out[16 + b] = (float)am;                      // Y_hat
    }
}

// ---------------------------------------------------------------------------
extern "C" void kernel_launch(void* const* d_in, const int* in_sizes, int n_in,
                              void* d_out, int out_size, void* d_ws, size_t ws_size,
                              hipStream_t stream) {
    (void)in_sizes; (void)n_in; (void)out_size; (void)ws_size;
    // setup_inputs() insertion order:
    const float* x      = (const float*)d_in[0];   // (2,4096,512)
    const float* mask   = (const float*)d_in[1];   // (2,4096)
    const float* fc1_w  = (const float*)d_in[2];   // (512,512)
    const float* fc1_b  = (const float*)d_in[3];
    const float* fnorm_w = (const float*)d_in[24];
    const float* fnorm_b = (const float*)d_in[25];
    const float* att_w1  = (const float*)d_in[26]; // (128,512)
    const float* att_b1  = (const float*)d_in[27];
    const float* att_w2  = (const float*)d_in[28]; // (1,128)
    const float* att_b2  = (const float*)d_in[29];
    const float* cls_w   = (const float*)d_in[30]; // (4,512)
    const float* cls_b   = (const float*)d_in[31];

    float* ws  = (float*)d_ws;
    float* out = (float*)d_out;

    // workspace layout (floats)
    const size_t off_h    = 0;                    // [NTOK,512]  running hidden
    const size_t off_ln   = 4194304;              // [NTOK,512]  LN out / out_proj out
    const size_t off_z    = 8388608;              // [NTOK,2048]
    const size_t off_xBC  = 25165824;             // [NTOK,2560] raw xBC / reused as yn
    const size_t off_dt   = 46137344;             // [NTOK,32]
    const size_t off_xBCc = 46399488;             // [NTOK,2560] conv'd xBC
    const size_t off_y    = 67371008;             // [NTOK,2048] scan out
    const size_t off_tp   = 84148224;             // [NTOK,128]  tanh proj
    const size_t off_a    = 85196800;             // [NTOK]      raw scores

    const dim3 gblk(32, 8);

    // fc1 + ReLU : h = relu(x @ fc1_w^T + b)
    wmma_gemm_kernel<1><<<dim3(1, NTOK / 16), gblk, 0, stream>>>(
        x, fc1_w, fc1_b, ws + off_h, NTOK, DMODEL, DMODEL);

    for (int l = 0; l < 2; ++l) {
        const float* ln_w     = (const float*)d_in[4 + 10 * l + 0];
        const float* ln_b     = (const float*)d_in[4 + 10 * l + 1];
        const float* in_proj  = (const float*)d_in[4 + 10 * l + 2];  // (4640,512)
        const float* conv_w   = (const float*)d_in[4 + 10 * l + 3];  // (2560,4)
        const float* conv_b   = (const float*)d_in[4 + 10 * l + 4];
        const float* dt_bias  = (const float*)d_in[4 + 10 * l + 5];
        const float* A_log    = (const float*)d_in[4 + 10 * l + 6];
        const float* Dp       = (const float*)d_in[4 + 10 * l + 7];
        const float* rms_w    = (const float*)d_in[4 + 10 * l + 8];
        const float* out_proj = (const float*)d_in[4 + 10 * l + 9];  // (512,2048)

        layernorm_kernel<<<NTOK, 256, 0, stream>>>(
            ws + off_h, ws + off_ln, ln_w, ln_b, DMODEL);

        // in_proj split into z / xBC / dt slices (rows of weight matrix)
        wmma_gemm_kernel<0><<<dim3(4, NTOK / 16), gblk, 0, stream>>>(
            ws + off_ln, in_proj, nullptr, ws + off_z, NTOK, DINNER, DMODEL);
        wmma_gemm_kernel<0><<<dim3(5, NTOK / 16), gblk, 0, stream>>>(
            ws + off_ln, in_proj + (size_t)DINNER * DMODEL, nullptr,
            ws + off_xBC, NTOK, DXBC, DMODEL);
        wmma_gemm_kernel<0><<<dim3(1, NTOK / 16), gblk, 0, stream>>>(
            ws + off_ln, in_proj + (size_t)(DINNER + DXBC) * DMODEL, nullptr,
            ws + off_dt, NTOK, NH, DMODEL);

        dwconv_silu_kernel<<<(NTOK * DXBC) / 256, 256, 0, stream>>>(
            ws + off_xBC, conv_w, conv_b, ws + off_xBCc, SEQL);

        scan_kernel<<<NB * NH, 256, 0, stream>>>(
            ws + off_dt, ws + off_xBCc, A_log, Dp, dt_bias, ws + off_y, SEQL);

        gated_rmsnorm_kernel<<<NTOK, 256, 0, stream>>>(
            ws + off_y, ws + off_z, rms_w, ws + off_xBC /* yn reuses xBC */);

        wmma_gemm_kernel<0><<<dim3(1, NTOK / 16), gblk, 0, stream>>>(
            ws + off_xBC, out_proj, nullptr, ws + off_ln /* o reuses ln */,
            NTOK, DMODEL, DINNER);

        residual_mask_kernel<<<(NTOK * DMODEL) / 256, 256, 0, stream>>>(
            ws + off_ln, ws + off_h, mask, DMODEL);
    }

    // final LN -> hf (in off_ln)
    layernorm_kernel<<<NTOK, 256, 0, stream>>>(
        ws + off_h, ws + off_ln, fnorm_w, fnorm_b, DMODEL);

    // attention: tanh(hf @ w1^T + b1) via WMMA, then dot w2
    wmma_gemm_kernel<2><<<dim3(1, NTOK / 16), gblk, 0, stream>>>(
        ws + off_ln, att_w1, att_b1, ws + off_tp, NTOK, 128, DMODEL);
    att_score_kernel<<<NTOK, 128, 0, stream>>>(
        ws + off_tp, att_w2, att_b2, ws + off_a);

    // masked softmax -> A written straight into d_out[18 .. 18+8192)
    masked_softmax_kernel<<<NB, 256, 0, stream>>>(
        ws + off_a, mask, out + 18, SEQL);

    // pooled h -> d_out[8210 .. 8210+1024)
    pool_kernel<<<4, 256, 0, stream>>>(out + 18, ws + off_ln, out + 8210, SEQL);

    // logits/probs/yhat -> d_out[0..18)
    head_kernel<<<1, 32, 0, stream>>>(out + 8210, cls_w, cls_b, out);
}
